// MemoryEfficientAttention_32942399160886
// MI455X (gfx1250) — compile-verified, matrix operand order checked by reading, not run on hardware
//
#include <hip/hip_runtime.h>
#include <hip/hip_bf16.h>

typedef __bf16 bf16;
typedef __attribute__((ext_vector_type(16))) __bf16 v16bf;
typedef __attribute__((ext_vector_type(8)))  float  v8f;

#define C_DIM 1024
#define T_DIM 8192
#define N_HEADS 16
#define HD 64
#define SCALE 0.125f

__device__ __forceinline__ v8f wmma_bf16(v16bf a, v16bf b, v8f c) {
  return __builtin_amdgcn_wmma_f32_16x16x32_bf16(
      /*neg_a=*/false, a, /*neg_b=*/false, b,
      /*c_mod=*/(short)0, c, /*reuse_a=*/false, /*reuse_b=*/false);
}

// Load one 16x32 bf16 WMMA A/B fragment from global memory (row-major,
// stride ld elements). Lane L in [0,16): row L, K {0..7, 16..23};
// lanes [16,32): row L-16, K {8..15, 24..31}.
#define LOAD_FRAG(dst, baseptr, ld)                                              \
  do {                                                                           \
    const bf16* _p = (baseptr) + (size_t)(lane & 15) * (size_t)(ld) +            \
                     ((lane & 16) ? 8 : 0);                                      \
    union { v16bf v; uint4 q[2]; } _u;                                           \
    _u.q[0] = *(const uint4*)_p;                                                 \
    _u.q[1] = *(const uint4*)(_p + 16);                                          \
    dst = _u.v;                                                                  \
  } while (0)

// Same fragment load but indexing a __shared__ array directly so the
// accesses lower to ds_load_b128.
#define LOAD_FRAG_LDS(dst, arr, baseoff, ld)                                     \
  do {                                                                           \
    const int _o = (baseoff) + (lane & 15) * (ld) + ((lane & 16) ? 8 : 0);       \
    union { v16bf v; uint4 q[2]; } _u;                                           \
    _u.q[0] = *(const uint4*)&(arr)[_o];                                         \
    _u.q[1] = *(const uint4*)&(arr)[_o + 16];                                    \
    dst = _u.v;                                                                  \
  } while (0)

// 16-byte async copy global -> LDS (gfx1250, tracked by ASYNCcnt).
__device__ __forceinline__ void async_copy_b128(unsigned lds_off, const bf16* g) {
  asm volatile("global_load_async_to_lds_b128 %0, %1, off"
               :: "v"(lds_off), "v"(g) : "memory");
}
#define WAIT_ASYNC(n) asm volatile("s_wait_asynccnt " #n ::: "memory")

// ---------------------------------------------------------------------------
// f32 -> bf16 conversion
// ---------------------------------------------------------------------------
__global__ void cvt_f32_bf16_kernel(const float* __restrict__ src,
                                    bf16* __restrict__ dst, long n) {
  long i = (long)blockIdx.x * blockDim.x + threadIdx.x;
  if (i < n) dst[i] = (bf16)src[i];
}

// ---------------------------------------------------------------------------
// GEMM: out[m,n] = sum_k A[m,k] * W[n,k] + bias[n]
// A: [32768 x 1024] bf16, W: [1024 x 1024] bf16.
// mode 0: bf16 out, [m,n]                       (K projection)
// mode 1: bf16 out, transposed-V layout         (V projection)
// mode 2: f32  out, [m,n]                       (final projection)
// mode 3: bf16 out, [m,n], scaled by SCALE      (Q projection)
//
// 256 threads = 8 waves (4 along M x 2 along N). WG tile 128x128, wave tile
// 32x64 (2x4 WMMA accumulators). K-step 32; A/W tiles double-buffered in LDS
// via global_load_async_to_lds_b128 (4 copies/thread/step, ASYNCcnt).
// ---------------------------------------------------------------------------
#define LDS_STRIDE 40                   // 32 + 8 pad (80B rows, conflict-free)
#define TILE_ELEMS (128 * LDS_STRIDE)   // 5120 elements per 128x32 tile

__global__ void gemm_nt_kernel(const bf16* __restrict__ A,
                               const bf16* __restrict__ W,
                               const float* __restrict__ bias,
                               void* __restrict__ out, int mode) {
  const int tid  = threadIdx.x;
  const int lane = tid & 31;
  const int wid  = tid >> 5;
  const int wm = wid & 3;        // 0..3  (M, 32 rows each)
  const int wn = wid >> 2;       // 0..1  (N, 64 cols each)
  const int mblk = blockIdx.y * 128;
  const int nblk = blockIdx.x * 128;

  __shared__ __align__(16) bf16 ldsA[2][TILE_ELEMS];
  __shared__ __align__(16) bf16 ldsW[2][TILE_ELEMS];

  // Copy slots: each 128x32 tile = 512 x16B chunks -> 2 chunks/thread/tile.
  // chunk c: row = c>>2 (0..127), col = (c&3)*8.
  const bf16* a_g[2];
  const bf16* w_g[2];
  int         t_off[2];
#pragma unroll
  for (int i = 0; i < 2; ++i) {
    const int c = tid * 2 + i, row = c >> 2, col = (c & 3) * 8;
    a_g[i]   = A + (size_t)(mblk + row) * C_DIM + col;
    w_g[i]   = W + (size_t)(nblk + row) * C_DIM + col;
    t_off[i] = row * LDS_STRIDE + col;
  }

#define ISSUE_TILE(buf, k)                                                     \
  do {                                                                         \
    _Pragma("unroll")                                                          \
    for (int _i = 0; _i < 2; ++_i) {                                           \
      async_copy_b128((unsigned)(size_t)&ldsA[buf][t_off[_i]], a_g[_i] + (k)); \
      async_copy_b128((unsigned)(size_t)&ldsW[buf][t_off[_i]], w_g[_i] + (k)); \
    }                                                                          \
  } while (0)

  ISSUE_TILE(0, 0);            // prologue

  v8f acc[2][4] = {};

#pragma unroll 1
  for (int ks = 0; ks < 32; ++ks) {
    const int cur = ks & 1;
    if (ks < 31) {
      ISSUE_TILE(cur ^ 1, (ks + 1) * 32);
      WAIT_ASYNC(4);           // current tile's 4 copies done; next 4 in flight
    } else {
      WAIT_ASYNC(0);
    }
    __syncthreads();

    v16bf wf[4];
#pragma unroll
    for (int i = 0; i < 4; ++i)
      LOAD_FRAG_LDS(wf[i], ldsW[cur], (wn * 64 + i * 16) * LDS_STRIDE, LDS_STRIDE);

#pragma unroll
    for (int mi = 0; mi < 2; ++mi) {
      v16bf af;
      LOAD_FRAG_LDS(af, ldsA[cur], (wm * 32 + mi * 16) * LDS_STRIDE, LDS_STRIDE);
#pragma unroll
      for (int ni = 0; ni < 4; ++ni)
        acc[mi][ni] = wmma_bf16(af, wf[ni], acc[mi][ni]);
    }

    __syncthreads();           // reads done before next iter overwrites buffer
  }

  // Epilogue. C layout: lane holds col n = nt + (lane&15); vgpr j holds
  // row m = mt + j + ((lane&16) ? 8 : 0).
  const int colL  = lane & 15;
  const int r_off = (lane & 16) ? 8 : 0;

#pragma unroll
  for (int mi = 0; mi < 2; ++mi) {
#pragma unroll
    for (int ni = 0; ni < 4; ++ni) {
      const int n = nblk + wn * 64 + ni * 16 + colL;
      const float bv = bias[n];
#pragma unroll
      for (int j = 0; j < 8; ++j) {
        const int m = mblk + wm * 32 + mi * 16 + j + r_off;
        float v = acc[mi][ni][j] + bv;
        if (mode == 0) {
          ((bf16*)out)[(size_t)m * C_DIM + n] = (bf16)v;
        } else if (mode == 3) {
          ((bf16*)out)[(size_t)m * C_DIM + n] = (bf16)(v * SCALE);
        } else if (mode == 2) {
          ((float*)out)[(size_t)m * C_DIM + n] = v;
        } else {
          // transposed V: b = m>>13, t = m&8191, h = n>>6, d = n&63
          const size_t row = (size_t)((m >> 13) * N_HEADS + (n >> 6)) * HD + (n & 63);
          ((bf16*)out)[row * T_DIM + (size_t)(m & 8191)] = (bf16)v;
        }
      }
    }
  }
}

// ---------------------------------------------------------------------------
// Block-local flash attention over 1024-token chunks. Q is pre-scaled by
// SCALE in its projection. grid: x = q-slice (0..7, 128 queries each),
// y = b*8 + chunk, z = head. 256 threads = 8 waves; each wave owns 16 query
// rows, iterates 1024 keys in 64-key tiles with online softmax.
// ---------------------------------------------------------------------------
__global__ void attn_kernel(const bf16* __restrict__ Q,
                            const bf16* __restrict__ K,
                            const bf16* __restrict__ VT,
                            bf16* __restrict__ O) {
  const int lane = threadIdx.x & 31;
  const int wave = threadIdx.x >> 5;       // 0..7
  const int qslice = blockIdx.x;           // 0..7
  const int bc = blockIdx.y;               // b*8 + chunk
  const int b = bc >> 3;
  const int cc = bc & 7;
  const int h = blockIdx.z;

  const int qrow0 = b * T_DIM + cc * 1024 + qslice * 128 + wave * 16;
  const int krow0 = b * T_DIM + cc * 1024;

  const bf16* qbase  = Q  + (size_t)qrow0 * C_DIM + h * HD;
  const bf16* kchunk = K  + (size_t)krow0 * C_DIM + h * HD;
  const bf16* vtbase = VT + (size_t)(b * N_HEADS + h) * HD * T_DIM + cc * 1024;

  __shared__ __align__(16) bf16 ldsP[8][16 * 72];   // per-wave P tile, padded

  v16bf qa0, qa1;
  LOAD_FRAG(qa0, qbase, C_DIM);
  LOAD_FRAG(qa1, qbase + 32, C_DIM);

  v8f o0 = {}, o1 = {}, o2 = {}, o3 = {};
  float rmax[8], rsum[8];
#pragma unroll
  for (int j = 0; j < 8; ++j) { rmax[j] = -3.0e38f; rsum[j] = 0.0f; }

  const int colL  = lane & 15;
  const int r_off = (lane & 16) ? 8 : 0;

  for (int kt = 0; kt < 16; ++kt) {
    const bf16* kb = kchunk + (size_t)(kt * 64) * C_DIM;

    // S = Q K^T for this 16x64 tile (Q already scaled)
    v8f s[4];
#pragma unroll
    for (int n = 0; n < 4; ++n) {
      v16bf b0, b1;
      LOAD_FRAG(b0, kb + (size_t)(n * 16) * C_DIM, C_DIM);
      LOAD_FRAG(b1, kb + (size_t)(n * 16) * C_DIM + 32, C_DIM);
      v8f z = {};
      z = wmma_bf16(qa0, b0, z);
      s[n] = wmma_bf16(qa1, b1, z);
    }

    // per-row tile max across the 16 lanes of each half
    float tm[8];
#pragma unroll
    for (int j = 0; j < 8; ++j)
      tm[j] = fmaxf(fmaxf(s[0][j], s[1][j]), fmaxf(s[2][j], s[3][j]));
#pragma unroll
    for (int j = 0; j < 8; ++j) {
      for (int off = 8; off >= 1; off >>= 1)
        tm[j] = fmaxf(tm[j], __shfl_xor(tm[j], off, 32));
    }

    // online-softmax rescale
    float f[8];
#pragma unroll
    for (int j = 0; j < 8; ++j) {
      float nm = fmaxf(rmax[j], tm[j]);
      f[j] = __expf(rmax[j] - nm);
      rmax[j] = nm;
    }
#pragma unroll
    for (int j = 0; j < 8; ++j) { o0[j] *= f[j]; o1[j] *= f[j]; o2[j] *= f[j]; o3[j] *= f[j]; }

    // P = exp(S - rowmax) -> bf16 in LDS (A-layout round trip); row sums
    float psum[8];
#pragma unroll
    for (int j = 0; j < 8; ++j) psum[j] = 0.0f;
#pragma unroll
    for (int n = 0; n < 4; ++n) {
#pragma unroll
      for (int j = 0; j < 8; ++j) {
        float p = __expf(s[n][j] - rmax[j]);
        psum[j] += p;
        ldsP[wave][(j + r_off) * 72 + n * 16 + colL] = (bf16)p;
      }
    }
#pragma unroll
    for (int j = 0; j < 8; ++j) {
      for (int off = 8; off >= 1; off >>= 1)
        psum[j] += __shfl_xor(psum[j], off, 32);
      rsum[j] = rsum[j] * f[j] + psum[j];
    }

    // reload P as A-fragments (keys 0..31, 32..63)
    v16bf pa0, pa1;
    LOAD_FRAG_LDS(pa0, ldsP[wave], 0, 72);
    LOAD_FRAG_LDS(pa1, ldsP[wave], 32, 72);

    // O += P * V  (V^T in memory: lane = d column, contiguous keys)
    const bf16* vt = vtbase + kt * 64;
    v16bf vb0, vb1;
    LOAD_FRAG(vb0, vt, T_DIM);
    LOAD_FRAG(vb1, vt + 32, T_DIM);
    o0 = wmma_bf16(pa0, vb0, o0); o0 = wmma_bf16(pa1, vb1, o0);
    LOAD_FRAG(vb0, vt + (size_t)16 * T_DIM, T_DIM);
    LOAD_FRAG(vb1, vt + (size_t)16 * T_DIM + 32, T_DIM);
    o1 = wmma_bf16(pa0, vb0, o1); o1 = wmma_bf16(pa1, vb1, o1);
    LOAD_FRAG(vb0, vt + (size_t)32 * T_DIM, T_DIM);
    LOAD_FRAG(vb1, vt + (size_t)32 * T_DIM + 32, T_DIM);
    o2 = wmma_bf16(pa0, vb0, o2); o2 = wmma_bf16(pa1, vb1, o2);
    LOAD_FRAG(vb0, vt + (size_t)48 * T_DIM, T_DIM);
    LOAD_FRAG(vb1, vt + (size_t)48 * T_DIM + 32, T_DIM);
    o3 = wmma_bf16(pa0, vb0, o3); o3 = wmma_bf16(pa1, vb1, o3);
  }

  // normalize and store (wave-coalesced b16 stores)
  bf16* obase = O + (size_t)qrow0 * C_DIM + h * HD;
  v8f os[4] = {o0, o1, o2, o3};
#pragma unroll
  for (int dt = 0; dt < 4; ++dt) {
#pragma unroll
    for (int j = 0; j < 8; ++j) {
      const float v = os[dt][j] / rsum[j];
      obase[(size_t)(j + r_off) * C_DIM + dt * 16 + colL] = (bf16)v;
    }
  }
}

// ---------------------------------------------------------------------------
extern "C" void kernel_launch(void* const* d_in, const int* in_sizes, int n_in,
                              void* d_out, int out_size, void* d_ws, size_t ws_size,
                              hipStream_t stream) {
  (void)in_sizes; (void)n_in; (void)out_size; (void)ws_size;
  const float* x  = (const float*)d_in[0];
  const float* wq = (const float*)d_in[1];
  const float* bq = (const float*)d_in[2];
  const float* wk = (const float*)d_in[3];
  const float* bk = (const float*)d_in[4];
  const float* wv = (const float*)d_in[5];
  const float* bv = (const float*)d_in[6];
  const float* wo = (const float*)d_in[7];
  const float* bo = (const float*)d_in[8];

  const long NTOK = 32768;                       // B*T
  const long NACT = NTOK * C_DIM;                // 33.5M elements
  const size_t SZ_ACT = (size_t)NACT * 2;        // bytes per bf16 activation buf

  char* ws = (char*)d_ws;
  bf16* xb  = (bf16*)(ws + 0 * SZ_ACT);
  bf16* qb  = (bf16*)(ws + 1 * SZ_ACT);
  bf16* kb  = (bf16*)(ws + 2 * SZ_ACT);
  bf16* vtb = (bf16*)(ws + 3 * SZ_ACT);
  bf16* ob  = (bf16*)(ws + 4 * SZ_ACT);
  bf16* wqb = (bf16*)(ws + 5 * SZ_ACT);
  bf16* wkb = wqb + (size_t)C_DIM * C_DIM;
  bf16* wvb = wkb + (size_t)C_DIM * C_DIM;
  bf16* wob = wvb + (size_t)C_DIM * C_DIM;

  const long NW = (long)C_DIM * C_DIM;
  cvt_f32_bf16_kernel<<<(NACT + 255) / 256, 256, 0, stream>>>(x,  xb,  NACT);
  cvt_f32_bf16_kernel<<<(NW   + 255) / 256, 256, 0, stream>>>(wq, wqb, NW);
  cvt_f32_bf16_kernel<<<(NW   + 255) / 256, 256, 0, stream>>>(wk, wkb, NW);
  cvt_f32_bf16_kernel<<<(NW   + 255) / 256, 256, 0, stream>>>(wv, wvb, NW);
  cvt_f32_bf16_kernel<<<(NW   + 255) / 256, 256, 0, stream>>>(wo, wob, NW);

  dim3 ggrid(C_DIM / 128, NTOK / 128, 1);     // (8, 256)
  gemm_nt_kernel<<<ggrid, 256, 0, stream>>>(xb, wqb, bq, qb,  3);  // Q (pre-scaled)
  gemm_nt_kernel<<<ggrid, 256, 0, stream>>>(xb, wkb, bk, kb,  0);  // K
  gemm_nt_kernel<<<ggrid, 256, 0, stream>>>(xb, wvb, bv, vtb, 1);  // V (transposed)

  dim3 agrid(8, 32, N_HEADS);                 // q-slices, b*chunks, heads
  attn_kernel<<<agrid, 256, 0, stream>>>(qb, kb, vtb, ob);

  gemm_nt_kernel<<<ggrid, 256, 0, stream>>>(ob, wob, bo, d_out, 2);
}
